// DecoderLSTM_43800076484802
// MI455X (gfx1250) — compile-verified, matrix operand order checked by reading, not run on hardware
//
#include <hip/hip_runtime.h>

typedef __attribute__((ext_vector_type(16))) __bf16 v16bf;
typedef __attribute__((ext_vector_type(8)))  __bf16 v8bf;
typedef __attribute__((ext_vector_type(8)))  float  v8f;
typedef __attribute__((ext_vector_type(4)))  float  v4f;

static constexpr int NB = 128;   // batch
static constexpr int HD = 256;   // hidden size
static constexpr int OC = 512;   // bilstm output channels (2H)

__device__ __forceinline__ __bf16 f2bf(float x) { return (__bf16)x; }
__device__ __forceinline__ float sigm(float x) { return 1.0f / (1.0f + __expf(-x)); }
// overflow-safe fast tanh: 1 - 2/(e^{2x}+1)  (e->inf => 1, e->0 => -1)
__device__ __forceinline__ float ftanh(float x) { return 1.0f - 2.0f / (__expf(2.0f * x) + 1.0f); }

__device__ __forceinline__ void loadA4(v16bf &f, int e, const float* __restrict__ p) {
  v4f v = *(const v4f*)p;
  f[e+0] = f2bf(v[0]); f[e+1] = f2bf(v[1]); f[e+2] = f2bf(v[2]); f[e+3] = f2bf(v[3]);
}
__device__ __forceinline__ void loadB8(v16bf &f, int e, const __bf16* __restrict__ p) {
  v8bf v = *(const v8bf*)p;
#pragma unroll
  for (int i = 0; i < 8; ++i) f[e+i] = v[i];
}

// ---------------------------------------------------------------------------
// Weight repack: fragment-order layout  dst[tile][kc][lane][16] (bf16) so a
// wave's WMMA B fragment is two contiguous 16B loads per lane at
// base + lane*32 + kc*1024 bytes.  Column n = tile*16 + lane%16,
// k = kc*32 + (lane/16)*8 + perm(e).  Source k<IN -> w0[n][k] (row stride IN),
// IN<=k<KT -> w1[n][k-IN] (row stride KT-IN), else 0.  Reads original f32.
// ---------------------------------------------------------------------------
__global__ void repack_frag_kernel(const float* __restrict__ w0, const float* __restrict__ w1,
                                   __bf16* __restrict__ dst, int IN, int KT, int NKC, int total)
{
  int i = blockIdx.x * 256 + threadIdx.x;
  if (i >= total) return;
  int fe = i & 511; int lane = fe >> 4; int e = fe & 15;
  int kc = (i >> 9) % NKC;
  int tile = i / (512 * NKC);
  int n = tile * 16 + (lane & 15);
  int k = kc * 32 + (lane >> 4) * 8 + ((e < 8) ? e : (e + 8));
  float v = 0.0f;
  if (k < IN)      v = w0[(size_t)n * IN + k];
  else if (k < KT) v = w1[(size_t)n * (KT - IN) + (k - IN)];
  dst[i] = (__bf16)v;
}

// ---------------------------------------------------------------------------
// Persistent per-layer bidirectional LSTM kernel.
// Grid: (8 batch-tiles of 16 rows, 2 directions). Block: 512 threads = 16
// waves (4 waves/SIMD for latency hiding). Wave w owns one 16-column subtile
// per gate: tile = g*16 + w, hidden columns n = w*16 + lane%16.  Per K-chunk
// the 4 B fragments are staged into a fresh register set (8 in-flight b128
// loads) and the next A fragment is prefetched before the 4-WMMA chain.
// c lives in registers; h round-trips through LDS.
// ---------------------------------------------------------------------------
template<int IN_DIM>
__global__ __launch_bounds__(512) void lstm_layer_kernel(
    const float* __restrict__ X, float* __restrict__ Y,
    const __bf16* __restrict__ wfrag,
    const float* __restrict__ bih, const float* __restrict__ bhh,
    const float* __restrict__ h0, const float* __restrict__ c0, int T)
{
  constexpr int KT  = IN_DIM + HD;
  constexpr int NKC = (KT + 31) / 32;
  constexpr int NKX = IN_DIM / 32;        // full X chunks (0 for IN=2,3)
  const int mb   = blockIdx.x;
  const int d    = blockIdx.y;
  const int w    = threadIdx.x >> 5;      // wave 0..15
  const int lane = threadIdx.x & 31;
  const int l15  = lane & 15;
  const int hi   = lane >> 4;

  const float* bihd = bih + d * 4 * HD;
  const float* bhhd = bhh + d * 4 * HD;

  __shared__ __bf16 hsm[16][HD + 8];      // 528B row stride: 16B aligned, banks skewed

  for (int idx = threadIdx.x; idx < 16 * HD; idx += 512) {
    int m = idx >> 8, n = idx & (HD - 1);
    hsm[m][n] = f2bf(h0[(size_t)d * NB * HD + (size_t)(mb * 16 + m) * HD + n]);
  }

  // hoisted per-gate fragment base pointers (invariant over time loop)
  const __bf16* fb[4];
#pragma unroll
  for (int g = 0; g < 4; ++g) {
    const int tile = g * 16 + w;
    fb[g] = wfrag + (size_t)d * 64 * NKC * 512 + (size_t)tile * NKC * 512 + (size_t)lane * 16;
  }

  const int n = w * 16 + l15;             // hidden column owned by this lane
  float creg[8];
  float bsum[4];
#pragma unroll
  for (int g = 0; g < 4; ++g) bsum[g] = bihd[g * HD + n] + bhhd[g * HD + n];
#pragma unroll
  for (int v = 0; v < 8; ++v) {
    int m = hi * 8 + v;
    creg[v] = c0[(size_t)d * NB * HD + (size_t)(mb * 16 + m) * HD + n];
  }
  __syncthreads();

  auto loadAH = [&](int kh) -> v16bf {
    const int kh0 = kh * 32 + hi * 8;
    v8bf h0v = *(const v8bf*)(&hsm[l15][kh0]);
    v8bf h1v = *(const v8bf*)(&hsm[l15][kh0 + 16]);
    v16bf a;
#pragma unroll
    for (int i = 0; i < 8; ++i) { a[i] = h0v[i]; a[8 + i] = h1v[i]; }
    return a;
  };

  for (int tt = 0; tt < T; ++tt) {
    const int t = d ? (T - 1 - tt) : tt;
    v8f acc[4] = {};

    if constexpr (IN_DIM % 32 == 0) {
      const float* pX = X + ((size_t)t * NB + mb * 16) * IN_DIM + (size_t)l15 * IN_DIM + hi * 8;
      auto loadAX = [&](int kc) -> v16bf {
        v16bf a;
        loadA4(a, 0,  pX + kc * 32);
        loadA4(a, 4,  pX + kc * 32 + 4);
        loadA4(a, 8,  pX + kc * 32 + 16);
        loadA4(a, 12, pX + kc * 32 + 20);
        return a;
      };
      v16bf a = loadAX(0);
      // ---- X phase ----
#pragma unroll 2
      for (int kc = 0; kc < NKX; ++kc) {
        v16bf bfs[4];
#pragma unroll
        for (int g = 0; g < 4; ++g) {
          const __bf16* p = fb[g] + (size_t)kc * 512;
          loadB8(bfs[g], 0, p); loadB8(bfs[g], 8, p + 8);
        }
        v16bf an = (kc + 1 < NKX) ? loadAX(kc + 1) : loadAH(0);
#pragma unroll
        for (int g = 0; g < 4; ++g)
          acc[g] = __builtin_amdgcn_wmma_f32_16x16x32_bf16(
              false, a, false, bfs[g], (short)0, acc[g], false, false);
        a = an;
      }
      // ---- H phase ----
#pragma unroll 2
      for (int kh = 0; kh < 8; ++kh) {
        v16bf bfs[4];
#pragma unroll
        for (int g = 0; g < 4; ++g) {
          const __bf16* p = fb[g] + (size_t)(NKX + kh) * 512;
          loadB8(bfs[g], 0, p); loadB8(bfs[g], 8, p + 8);
        }
        v16bf an = (kh + 1 < 8) ? loadAH(kh + 1) : a;
#pragma unroll
        for (int g = 0; g < 4; ++g)
          acc[g] = __builtin_amdgcn_wmma_f32_16x16x32_bf16(
              false, a, false, bfs[g], (short)0, acc[g], false, false);
        a = an;
      }
    } else {
      // ---- generic phase for tiny IN (2,3): element-wise A, frag B ----
      const float* Xrow = X + ((size_t)t * NB + mb * 16) * IN_DIM;
      for (int kc = 0; kc < NKC; ++kc) {
        v16bf bfs[4];
#pragma unroll
        for (int g = 0; g < 4; ++g) {
          const __bf16* p = fb[g] + (size_t)kc * 512;
          loadB8(bfs[g], 0, p); loadB8(bfs[g], 8, p + 8);
        }
        v16bf a;
#pragma unroll
        for (int e = 0; e < 16; ++e) {
          int k = kc * 32 + hi * 8 + ((e < 8) ? e : (e + 8));
          __bf16 v = (__bf16)0.0f;
          if (k < IN_DIM)  v = f2bf(Xrow[(size_t)l15 * IN_DIM + k]);
          else if (k < KT) v = hsm[l15][k - IN_DIM];
          a[e] = v;
        }
#pragma unroll
        for (int g = 0; g < 4; ++g)
          acc[g] = __builtin_amdgcn_wmma_f32_16x16x32_bf16(
              false, a, false, bfs[g], (short)0, acc[g], false, false);
      }
    }

    __syncthreads();   // all waves done reading hsm

#pragma unroll
    for (int v = 0; v < 8; ++v) {
      const int m = hi * 8 + v;
      float iv = sigm (acc[0][v] + bsum[0]);
      float fv = sigm (acc[1][v] + bsum[1]);
      float gv = ftanh(acc[2][v] + bsum[2]);
      float ov = sigm (acc[3][v] + bsum[3]);
      float cc = fv * creg[v] + iv * gv;
      float hh = ov * ftanh(cc);
      creg[v] = cc;
      hsm[m][n] = f2bf(hh);
      Y[((size_t)t * NB + mb * 16 + m) * OC + (size_t)d * HD + n] = hh;
    }
    __syncthreads();
  }
}

// ---------------------------------------------------------------------------
// Mid conv1d (same padding, tanh) as WMMA GEMM over K = cin*ksz, im2col via
// index math from an LDS-staged input tile; fragment-packed weights with a
// 2-deep B double-buffer. Grid: (T/16, B, cout/128).
// ---------------------------------------------------------------------------
template<int KSZ>
__global__ __launch_bounds__(256) void conv_wmma_kernel(
    const float* __restrict__ X, float* __restrict__ Yc,
    const __bf16* __restrict__ Wf, const float* __restrict__ bias,
    int T, int cin, int cout)
{
  constexpr int PAD  = (KSZ - 1) / 2;
  constexpr int ROWS = 16 + 2 * PAD;
  __shared__ __bf16 xs[22 * 512];
  const int t0   = blockIdx.x * 16;
  const int b    = blockIdx.y;
  const int w    = threadIdx.x >> 5;
  const int lane = threadIdx.x & 31;
  const int l15  = lane & 15, hi = lane >> 4;

  for (int idx = threadIdx.x; idx < ROWS * cin; idx += 256) {
    int r = idx / cin, ci = idx - r * cin;
    int ts = t0 - PAD + r;
    float v = (ts >= 0 && ts < T) ? X[((size_t)ts * NB + b) * cin + ci] : 0.0f;
    xs[idx] = f2bf(v);
  }
  __syncthreads();

  const int co0 = blockIdx.z * 128 + w * 16;
  if (co0 >= cout) return;
  const int CK = cin * KSZ;
  const int NKC = CK / 32;
  const __bf16* fbc = Wf + (size_t)(co0 >> 4) * NKC * 512 + (size_t)lane * 16;
  v8f acc = {};
  v16bf bcur;
  loadB8(bcur, 0, fbc); loadB8(bcur, 8, fbc + 8);
  for (int kc = 0; kc < NKC; ++kc) {
    v16bf bnext = bcur;
    if (kc + 1 < NKC) {
      const __bf16* p = fbc + (size_t)(kc + 1) * 512;
      loadB8(bnext, 0, p); loadB8(bnext, 8, p + 8);
    }
    v16bf a;
#pragma unroll
    for (int e = 0; e < 16; ++e) {
      int kidx = kc * 32 + hi * 8 + ((e < 8) ? e : (e + 8));
      int ci = kidx / KSZ, kk = kidx - ci * KSZ;
      a[e] = xs[(l15 + kk) * cin + ci];
    }
    acc = __builtin_amdgcn_wmma_f32_16x16x32_bf16(
        false, a, false, bcur, (short)0, acc, false, false);
    bcur = bnext;
  }
  const int co = co0 + l15;
  const float bb = bias[co];
#pragma unroll
  for (int v = 0; v < 8; ++v) {
    int t = t0 + hi * 8 + v;
    if (t < T) Yc[((size_t)t * NB + b) * cout + co] = ftanh(acc[v] + bb);
  }
}

// Final conv (cout == 1): trivial VALU kernel, no tanh.
__global__ void conv_final_kernel(const float* __restrict__ X, float* __restrict__ out,
                                  const float* __restrict__ W, const float* __restrict__ bias,
                                  int T, int cin, int ksz)
{
  int idx = blockIdx.x * 256 + threadIdx.x;
  if (idx >= NB * T) return;
  int b = idx / T, t = idx - b * T;
  int pad = (ksz - 1) / 2;
  float acc = bias[0];
  for (int kk = 0; kk < ksz; ++kk) {
    int ts = t + kk - pad;
    if (ts < 0 || ts >= T) continue;
    const float* xp = X + ((size_t)ts * NB + b) * cin;
    for (int ci = 0; ci < cin; ++ci) acc += W[ci * ksz + kk] * xp[ci];
  }
  out[(size_t)b * T + t] = acc;
}

// ------------------------------- utilities ---------------------------------
__global__ void add_kernel(float* __restrict__ dst, const float* __restrict__ src, int n) {
  int i = blockIdx.x * 256 + threadIdx.x;
  if (i < n) dst[i] += src[i];
}
__global__ void repeat_kernel(float* __restrict__ dst, const float* __restrict__ src,
                              int Tdst, int factor) {
  int n = Tdst * NB * OC;
  int i = blockIdx.x * 256 + threadIdx.x;
  if (i >= n) return;
  int t = i / (NB * OC);
  int rem = i - t * (NB * OC);
  dst[i] = src[(size_t)(t / factor) * (NB * OC) + rem];
}
__global__ void build_bb_kernel(const float* __restrict__ x1, float* __restrict__ dst) {
  int i = blockIdx.x * 256 + threadIdx.x;               // 8*128*2
  if (i >= 8 * NB * 2) return;
  int t = i / (NB * 2); int r = i - t * (NB * 2); int b = r >> 1; int c = r & 1;
  dst[i] = x1[((size_t)b * 288 + t * 36) * 2 + c];
}
__global__ void build_in5_kernel(const float* __restrict__ x1, const float* __restrict__ x3,
                                 float* __restrict__ dst) {
  int i = blockIdx.x * 256 + threadIdx.x;               // 288*128*3
  if (i >= 288 * NB * 3) return;
  int t = i / (NB * 3); int r = i - t * (NB * 3); int b = r / 3; int c = r - b * 3;
  float v;
  if (c == 0) v = x3[((size_t)b * 288 + (287 - t)) * 2 + 0];
  else        v = x1[((size_t)b * 288 + t) * 2 + (c - 1)];
  dst[i] = v;
}

// ------------------------------ host driver --------------------------------
extern "C" void kernel_launch(void* const* d_in, const int* in_sizes, int n_in,
                              void* d_out, int out_size, void* d_ws, size_t ws_size,
                              hipStream_t stream) {
  (void)in_sizes; (void)n_in; (void)out_size; (void)ws_size;
  const float* x1    = (const float*)d_in[0];
  const float* x3    = (const float*)d_in[1];
  const float* enc_h = (const float*)d_in[2];
  const float* enc_c = (const float*)d_in[3];
  // params flattened in sorted-key pytree order:
  // cnn00:4-7  cnn01:8-11  cnn02:12-15  cnn03:16-21  cnn04:22-25
  // lstm00:26-30  lstm01:31-35  lstm03:36-40  lstm04:41-45  lstm05:46-50
  // (per-lstm order: bhh, bih, whh, wih, wih0)

  float* outF = (float*)d_out;
  float* out0 = outF + 0;
  float* out1 = outF + 1024;
  float* out2 = outF + 5120;
  float* out3 = outF + 41984;
  float* out4 = outF + 54272;

  const size_t SEQ = (size_t)288 * NB * OC;
  const size_t CVN = (size_t)288 * NB * 256;
  float* S0  = (float*)d_ws;
  float* S1  = S0 + SEQ;
  float* S2  = S1 + SEQ;
  float* CV0 = S2 + SEQ;
  float* CV1 = CV0 + CVN;
  __bf16* pool = (__bf16*)(CV1 + CVN);
  size_t poff = 0;

  auto repack = [&](const float* w0, const float* w1, __bf16* dst, int IN, int KT, int nTiles) {
    int NKC = (KT + 31) / 32;
    int total = nTiles * NKC * 512;
    repack_frag_kernel<<<dim3((total + 255) / 256), dim3(256), 0, stream>>>(
        w0, w1, dst, IN, KT, NKC, total);
  };

  struct LP { const float *bhh, *bih; const __bf16* frag[4]; int nl, in0; };
  auto mkLstm = [&](int base, int nl, int in0) -> LP {
    LP p;
    p.bhh = (const float*)d_in[base + 0];
    p.bih = (const float*)d_in[base + 1];
    const float* whh  = (const float*)d_in[base + 2];
    const float* wih  = (const float*)d_in[base + 3];
    const float* wih0 = (const float*)d_in[base + 4];
    for (int l = 0; l < nl; ++l) {
      const int IN = (l == 0) ? in0 : 512;
      const int KT = IN + 256;
      const int NKC = (KT + 31) / 32;
      __bf16* dst = pool + poff;
      poff += (size_t)2 * 64 * NKC * 512;
      for (int dd = 0; dd < 2; ++dd) {
        const float* w0 = (l == 0) ? (wih0 + (size_t)dd * 1024 * in0)
                                   : (wih + ((size_t)(l - 1) * 2 + dd) * 1024 * 512);
        const float* w1 = whh + ((size_t)l * 2 + dd) * 1024 * 256;
        repack(w0, w1, dst + (size_t)dd * 64 * NKC * 512, IN, KT, 64);
      }
      p.frag[l] = dst;
    }
    p.nl = nl; p.in0 = in0;
    return p;
  };
  LP L00 = mkLstm(26, 2, 2);
  LP L01 = mkLstm(31, 2, 512);
  LP L03 = mkLstm(36, 4, 512);
  LP L04 = mkLstm(41, 4, 512);
  LP L05 = mkLstm(46, 4, 3);

  auto mkConv = [&](const float* wsrc, int cout, int cin, int ksz) -> const __bf16* {
    int CK = cin * ksz;
    int nT = cout / 16;
    __bf16* dst = pool + poff;
    poff += (size_t)nT * (CK / 32) * 512;
    repack(wsrc, wsrc, dst, CK, CK, nT);
    return dst;
  };
  const __bf16* W00 = mkConv((const float*)d_in[4],  64,  512, 3);
  const __bf16* W01 = mkConv((const float*)d_in[8],  64,  512, 3);
  const __bf16* W02 = mkConv((const float*)d_in[12], 256, 512, 5);
  const __bf16* W3a = mkConv((const float*)d_in[16], 128, 512, 3);
  const __bf16* W3b = mkConv((const float*)d_in[18], 64,  128, 5);
  const __bf16* W04 = mkConv((const float*)d_in[22], 256, 512, 5);

  auto lstm_layer = [&](int in_dim, int T, const float* X, float* Y, const __bf16* wf,
                        const float* bih, const float* bhh,
                        const float* h0, const float* c0) {
    dim3 g(8, 2), blk(512);
    if (in_dim == 512)
      lstm_layer_kernel<512><<<g, blk, 0, stream>>>(X, Y, wf, bih, bhh, h0, c0, T);
    else if (in_dim == 2)
      lstm_layer_kernel<2><<<g, blk, 0, stream>>>(X, Y, wf, bih, bhh, h0, c0, T);
    else
      lstm_layer_kernel<3><<<g, blk, 0, stream>>>(X, Y, wf, bih, bhh, h0, c0, T);
  };

  auto bilstm = [&](const LP& P, int T, int enc_k, const float* Xin) -> float* {
    const float* src = Xin;
    float* dst = S0;
    float* outp = nullptr;
    for (int l = 0; l < P.nl; ++l) {
      int in_dim = (l == 0) ? P.in0 : 512;
      const float* bih = P.bih + (size_t)l * 2048;
      const float* bhh = P.bhh + (size_t)l * 2048;
      const float* h0 = enc_h + ((size_t)enc_k * 8 + 2 * l) * NB * HD;
      const float* c0 = enc_c + ((size_t)enc_k * 8 + 2 * l) * NB * HD;
      lstm_layer(in_dim, T, src, dst, P.frag[l], bih, bhh, h0, c0);
      outp = dst; src = dst; dst = (dst == S0) ? S1 : S0;
    }
    return outp;
  };

  auto conv_mid = [&](const float* X, float* Yc, const __bf16* Wf, const float* bias,
                      int T, int cin, int cout, int ksz) {
    dim3 g((T + 15) / 16, NB, (cout + 127) / 128), blk(256);
    if (ksz == 3) conv_wmma_kernel<3><<<g, blk, 0, stream>>>(X, Yc, Wf, bias, T, cin, cout);
    else          conv_wmma_kernel<5><<<g, blk, 0, stream>>>(X, Yc, Wf, bias, T, cin, cout);
  };
  auto conv_fin = [&](const float* X, float* o, const float* W, const float* bias,
                      int T, int cin, int ksz) {
    int n = NB * T;
    conv_final_kernel<<<dim3((n + 255) / 256), dim3(256), 0, stream>>>(X, o, W, bias, T, cin, ksz);
  };
  auto addv = [&](float* dst, const float* src, int n) {
    add_kernel<<<dim3((n + 255) / 256), dim3(256), 0, stream>>>(dst, src, n);
  };
  auto rep = [&](float* dst, const float* src, int Tdst, int f) {
    int n = Tdst * NB * OC;
    repeat_kernel<<<dim3((n + 255) / 256), dim3(256), 0, stream>>>(dst, src, Tdst, f);
  };

  // ---- Stage A: lstm00 (T=8, in=2) + cnn00 -> out0 ----
  build_bb_kernel<<<dim3(8), dim3(256), 0, stream>>>(x1, S2);
  float* oA = bilstm(L00, 8, 0, S2);
  conv_mid(oA, CV0, W00, (const float*)d_in[5], 8, 512, 64, 3);
  conv_fin(CV0, out0, (const float*)d_in[6], (const float*)d_in[7], 8, 64, 5);
  rep(S2, oA, 32, 4);

  // ---- Stage B: lstm01 (T=32) + residual + cnn01 -> out1 ----
  float* oB = bilstm(L01, 32, 1, S2);
  addv(oB, S2, 32 * NB * OC);
  conv_mid(oB, CV0, W01, (const float*)d_in[9], 32, 512, 64, 3);
  conv_fin(CV0, out1, (const float*)d_in[10], (const float*)d_in[11], 32, 64, 5);
  rep(S2, oB, 96, 6);

  // ---- Stage C: lstm03 (T=96, 4 layers) + residual + cnn03 -> out3 ----
  float* oC = bilstm(L03, 96, 3, S2);
  addv(oC, S2, 96 * NB * OC);
  conv_mid(oC, CV0, W3a, (const float*)d_in[17], 96, 512, 128, 3);
  conv_mid(CV0, CV1, W3b, (const float*)d_in[19], 96, 128, 64, 5);
  conv_fin(CV1, out3, (const float*)d_in[20], (const float*)d_in[21], 96, 64, 7);
  rep(S2, oC, 288, 3);

  // ---- Stage D: lstm04 (T=288, 4 layers) + residual + cnn04 -> out4 ----
  float* oD = bilstm(L04, 288, 4, S2);
  addv(oD, S2, 288 * NB * OC);
  conv_mid(oD, CV0, W04, (const float*)d_in[23], 288, 512, 256, 5);
  conv_fin(CV0, out4, (const float*)d_in[24], (const float*)d_in[25], 288, 256, 7);

  // ---- Stage E: lstm05 (T=288, in=3, 4 layers) + cnn02 -> out2; out4 += out2 ----
  build_in5_kernel<<<dim3((288 * NB * 3 + 255) / 256), dim3(256), 0, stream>>>(x1, x3, S2);
  float* oE = bilstm(L05, 288, 5, S2);
  conv_mid(oE, CV0, W02, (const float*)d_in[13], 288, 512, 256, 5);
  conv_fin(CV0, out2, (const float*)d_in[14], (const float*)d_in[15], 288, 256, 7);
  addv(out4, out2, 288 * NB);
}
